// VQVAELayer_44547400794157
// MI455X (gfx1250) — matrix operand order repaired and checked by reading, not compiled, not run on hardware
//
#include <hip/hip_runtime.h>
#include <hip/hip_bf16.h>

// VQ-VAE vector quantization for MI455X (gfx1250, wave32, WMMA).
//
// x: [65536, 64] fp32, w: [64, 4096] fp32.
// score(n, j) = x_n . w_j - 0.5*||w_j||^2 ; argmax_j ; gather w[:, argmax].
//
// v_wmma_f32_16x16x32_bf16 with bf16 hi/lo split of both operands
// (hh + lh + hl), fp32 accumulate, -0.5*||w||^2 preloaded into the
// accumulator (pre-negated in the prepass so no per-tile VALU negation).
// Codes ride the A/M axis, pixels the B/N axis, so each pixel's candidate
// scores land inside one lane and the running argmax is branch-free per-lane
// VALU that co-executes with the XDL matrix ops.
// Code tiles are staged through double-buffered LDS with
// GLOBAL_LOAD_ASYNC_TO_LDS_B128 (ASYNCcnt) when the builtin is available.

typedef __bf16 bf16;
typedef bf16  v16bf __attribute__((ext_vector_type(16)));
typedef float v8f   __attribute__((ext_vector_type(8)));
typedef int   v4i   __attribute__((ext_vector_type(4)));

#define NPIX       65536
#define DIM        64
#define NCODE      4096
#define NTILE      256            // 4096 / 16 code tiles
#define WSQ_OFF_DW 262144         // wsA occupies 1 MB = 262144 dwords of d_ws

#define WMMA_BF16(a, b, c) \
  __builtin_amdgcn_wmma_f32_16x16x32_bf16(false, (a), false, (b), (short)0, (c), false, false)

#if defined(__has_builtin)
#if __has_builtin(__builtin_amdgcn_global_load_async_to_lds_b128)
#define VQ_ASYNC_LDS 1
#endif
#endif

__device__ __forceinline__ void vq_wait_async0() {
#if defined(__has_builtin) && __has_builtin(__builtin_amdgcn_s_wait_asynccnt)
  __builtin_amdgcn_s_wait_asynccnt(0);
#else
  asm volatile("s_wait_asynccnt 0x0" ::: "memory");
#endif
}

#ifdef VQ_ASYNC_LDS
__device__ __forceinline__ void vq_stage16_async(const uint4* g, uint4* l) {
  // Cache -> LDS DMA, no VGPR round-trip, tracked on ASYNCcnt.
  // Builtin signature (per compiler diagnostic): (v4i*, v4i*, int offset, int cpol)
  __builtin_amdgcn_global_load_async_to_lds_b128((v4i*)g, (v4i*)l,
                                                 /*offset=*/0, /*cpol=*/0);
}
#endif

// ---------------------------------------------------------------------------
// Prepass 1: wsq[k] = -0.5 * sum_d w[d,k]^2  (pre-negated accumulator bias)
// ---------------------------------------------------------------------------
__global__ void vq_wsq_kernel(const float* __restrict__ w, float* __restrict__ wsq) {
  const int k = blockIdx.x * blockDim.x + threadIdx.x;       // 0..4095
  float s = 0.0f;
#pragma unroll
  for (int d = 0; d < DIM; ++d) {
    const float v = w[d * NCODE + k];
    s += v * v;                                               // coalesced over k
  }
  wsq[k] = -0.5f * s;
}

// ---------------------------------------------------------------------------
// Prepass 2: swizzle w into WMMA A-fragment layout, bf16 hi/lo planes.
// Layout: wsA[tile][kstep(2)][plane(2: hi,lo)][lane(32)][v(8)] dwords,
// matching the 16-bit 16x32 A-matrix VGPR layout (ISA 7.12.2).
// One thread per (tile, kstep, lane, v) = 131072 threads.
// ---------------------------------------------------------------------------
__global__ void vq_swizzle_kernel(const float* __restrict__ w, unsigned* __restrict__ wsA) {
  const unsigned gid  = blockIdx.x * blockDim.x + threadIdx.x;
  const unsigned v    = gid & 7u;
  const unsigned lane = (gid >> 3) & 31u;
  const unsigned h    = (gid >> 8) & 1u;                      // kstep
  const unsigned tile = gid >> 9;
  const unsigned m    = lane & 15u;                           // code row in tile
  const unsigned q    = lane >> 4;
  // A-matrix 16x32 (16-bit): lanes0-15 V0-3 hold K0-7, lanes16-31 V0-3 K8-15,
  //                          lanes0-15 V4-7 K16-23, lanes16-31 V4-7 K24-31.
  const unsigned klocal = (v < 4u) ? (2u * v + 8u * q) : (16u + 2u * (v - 4u) + 8u * q);
  const unsigned d    = 32u * h + klocal;
  const unsigned code = tile * 16u + m;

  const float f0 = w[d * NCODE + code];
  const float f1 = w[(d + 1u) * NCODE + code];
  const bf16 h0 = (bf16)f0, h1 = (bf16)f1;
  const bf16 l0 = (bf16)(f0 - (float)h0), l1 = (bf16)(f1 - (float)h1);

  union { bf16 b[2]; unsigned u; } pk;
  const unsigned base = ((tile * 2u + h) * 2u) * 256u + lane * 8u + v;
  pk.b[0] = h0; pk.b[1] = h1; wsA[base]        = pk.u;        // hi plane
  pk.b[0] = l0; pk.b[1] = l1; wsA[base + 256u] = pk.u;        // lo plane
}

// ---------------------------------------------------------------------------
// Main fused GEMM + argmax + gather. 256 WGs x 256 threads (8 wave32).
// Each wave owns 32 pixels (two 16-row slabs, B-fragments register-resident);
// the 4 KB code tile is double-buffered through LDS and shared by all waves.
// ---------------------------------------------------------------------------
__global__ __launch_bounds__(256) void vq_main_kernel(
    const float*    __restrict__ x,
    const float*    __restrict__ w,
    const unsigned* __restrict__ wsA,
    const float*    __restrict__ wsq,
    float*          __restrict__ outq,
    int*            __restrict__ outi) {
  __shared__ uint4 lds4[512];                                  // 2 x 4 KB tiles
  unsigned* lds = (unsigned*)lds4;

  const int tid  = threadIdx.x;
  const int lane = tid & 31;
  const int wave = tid >> 5;
  const int n    = lane & 15;                                  // pixel within slab
  const int q    = lane >> 4;
  const int rbase0 = (int)blockIdx.x * 256 + wave * 32;
  const int rbase1 = rbase0 + 16;

  // ---- Build x B-fragments (bf16 hi/lo), resident across the whole K loop.
  // Dense 32x16 B layout: lane l (n=l&15, q=l>>4), elems 0-7 K=8q+s, elems 8-15 K=16+8q+(s-8).
  v16bf Bh[2][2], Bl[2][2];                                    // [slab][kstep]
#pragma unroll
  for (int s = 0; s < 2; ++s) {
    const float* xr = x + (size_t)((s ? rbase1 : rbase0) + n) * DIM;
#pragma unroll
    for (int h = 0; h < 2; ++h) {
      const float4* p4 = (const float4*)(xr + 32 * h + 8 * q);
      float fx[16];
      *(float4*)&fx[0]  = p4[0];
      *(float4*)&fx[4]  = p4[1];
      *(float4*)&fx[8]  = p4[4];
      *(float4*)&fx[12] = p4[5];
      v16bf bh, bl;
#pragma unroll
      for (int j = 0; j < 16; ++j) {
        const float f  = fx[j];
        const bf16 hi = (bf16)f;
        bh[j] = hi;
        bl[j] = (bf16)(f - (float)hi);
      }
      Bh[s][h] = bh;
      Bl[s][h] = bl;
    }
  }

  const uint4* gA = (const uint4*)wsA;                         // 256 uint4 / tile

  // ---- Stage code tile 0 into LDS buffer 0.
#ifdef VQ_ASYNC_LDS
  vq_stage16_async(gA + tid, lds4 + tid);
  vq_wait_async0();
#else
  lds4[tid] = gA[tid];
#endif
  __syncthreads();

  float best0 = -3.4e38f, best1 = -3.4e38f;
  int   bidx0 = 0,        bidx1 = 0;

  for (int t = 0; t < NTILE; ++t) {
    const bool more = (t + 1) < NTILE;

#ifdef VQ_ASYNC_LDS
    if (more)                                                  // DMA next tile into
      vq_stage16_async(gA + (t + 1) * 256 + tid,               // the other buffer
                       lds4 + ((t + 1) & 1) * 256 + tid);
#else
    uint4 nxt = {0u, 0u, 0u, 0u};
    if (more) nxt = gA[(t + 1) * 256 + tid];                   // prefetch next tile
#endif

    const unsigned* buf = lds + (t & 1) * 1024;
    const v16bf A0 = *(const v16bf*)(buf + (0 * 32 + lane) * 8);  // k0 hi
    const v16bf A1 = *(const v16bf*)(buf + (1 * 32 + lane) * 8);  // k0 lo
    const v16bf A2 = *(const v16bf*)(buf + (2 * 32 + lane) * 8);  // k1 hi
    const v16bf A3 = *(const v16bf*)(buf + (3 * 32 + lane) * 8);  // k1 lo

    // Accumulator preloaded with -0.5*||w||^2 for this lane's 8 codes
    // (already negated in the prepass -> plain loads, no VALU).
    const float* wqp = wsq + t * 16 + q * 8;
    const float4 w0 = *(const float4*)wqp;
    const float4 w1 = *(const float4*)(wqp + 4);
    v8f c0;
    c0[0] = w0.x; c0[1] = w0.y; c0[2] = w0.z; c0[3] = w0.w;
    c0[4] = w1.x; c0[5] = w1.y; c0[6] = w1.z; c0[7] = w1.w;
    v8f c1 = c0;

    // bf16x2 emulated-fp32 GEMM: hh + lh + hl over both 32-wide K steps.
    c0 = WMMA_BF16(A0, Bh[0][0], c0);
    c0 = WMMA_BF16(A1, Bh[0][0], c0);
    c0 = WMMA_BF16(A0, Bl[0][0], c0);
    c0 = WMMA_BF16(A2, Bh[0][1], c0);
    c0 = WMMA_BF16(A3, Bh[0][1], c0);
    c0 = WMMA_BF16(A2, Bl[0][1], c0);

    c1 = WMMA_BF16(A0, Bh[1][0], c1);
    c1 = WMMA_BF16(A1, Bh[1][0], c1);
    c1 = WMMA_BF16(A0, Bl[1][0], c1);
    c1 = WMMA_BF16(A2, Bh[1][1], c1);
    c1 = WMMA_BF16(A3, Bh[1][1], c1);
    c1 = WMMA_BF16(A2, Bl[1][1], c1);

    // Per-lane running argmax over this lane's 8 codes (co-executes with XDL).
    const int mb = t * 16 + q * 8;
#pragma unroll
    for (int i = 0; i < 8; ++i) {
      const float s0 = c0[i];
      const bool p0 = s0 > best0;
      best0 = p0 ? s0 : best0;
      bidx0 = p0 ? (mb + i) : bidx0;
      const float s1 = c1[i];
      const bool p1 = s1 > best1;
      best1 = p1 ? s1 : best1;
      bidx1 = p1 ? (mb + i) : bidx1;
    }

#ifdef VQ_ASYNC_LDS
    vq_wait_async0();                                          // DMA done before
#else
    if (more) lds4[((t + 1) & 1) * 256 + tid] = nxt;           // publish buffer
#endif
    __syncthreads();
  }

  // ---- Merge lane n with lane n+16 (codes 0-7 vs 8-15 of every tile).
  {
    const float os0 = __shfl_xor(best0, 16, 32);
    const int   oi0 = __shfl_xor(bidx0, 16, 32);
    const bool t0 = (os0 > best0) || (os0 == best0 && oi0 < bidx0);
    best0 = t0 ? os0 : best0;  bidx0 = t0 ? oi0 : bidx0;
    const float os1 = __shfl_xor(best1, 16, 32);
    const int   oi1 = __shfl_xor(bidx1, 16, 32);
    const bool t1 = (os1 > best1) || (os1 == best1 && oi1 < bidx1);
    best1 = t1 ? os1 : best1;  bidx1 = t1 ? oi1 : bidx1;
  }

  if (lane < 16) {
    outi[rbase0 + n] = bidx0;
    outi[rbase1 + n] = bidx1;
  }

  // ---- Gather winning codewords (w is L2-resident): 64 floats/row, 32 lanes.
#pragma unroll 4
  for (int r = 0; r < 16; ++r) {
    const int id0 = __shfl(bidx0, r, 32);
    const int id1 = __shfl(bidx1, r, 32);
    const float a0 = w[(size_t)lane * NCODE + id0];
    const float a1 = w[(size_t)(lane + 32) * NCODE + id0];
    const float b0 = w[(size_t)lane * NCODE + id1];
    const float b1 = w[(size_t)(lane + 32) * NCODE + id1];
    outq[(size_t)(rbase0 + r) * DIM + lane]      = a0;
    outq[(size_t)(rbase0 + r) * DIM + lane + 32] = a1;
    outq[(size_t)(rbase1 + r) * DIM + lane]      = b0;
    outq[(size_t)(rbase1 + r) * DIM + lane + 32] = b1;
  }
}

// ---------------------------------------------------------------------------
extern "C" void kernel_launch(void* const* d_in, const int* in_sizes, int n_in,
                              void* d_out, int out_size, void* d_ws, size_t ws_size,
                              hipStream_t stream) {
  const float* x = (const float*)d_in[0];   // [64,32,32,64] fp32
  const float* w = (const float*)d_in[1];   // [64,4096] fp32

  unsigned* wsA = (unsigned*)d_ws;                          // 1 MB swizzled codebook
  float*    wsq = (float*)((char*)d_ws + WSQ_OFF_DW * 4);   // 16 KB -0.5*||w||^2

  float* outq = (float*)d_out;                              // quantized [N,64]
  int*   outi = (int*)d_out + (size_t)NPIX * DIM;           // indices [N]

  vq_wsq_kernel<<<NCODE / 256, 256, 0, stream>>>(w, wsq);
  vq_swizzle_kernel<<<(NTILE * 2 * 32 * 8) / 256, 256, 0, stream>>>(w, wsA);
  vq_main_kernel<<<NPIX / 256, 256, 0, stream>>>(x, w, wsA, wsq, outq, outi);
}
